// ControllableNCA_65094524338956
// MI455X (gfx1250) — matrix operand order repaired and checked by reading, not compiled, not run on hardware
//
#include <hip/hip_runtime.h>

typedef _Float16 v16h __attribute__((ext_vector_type(16)));
typedef _Float16 v8h  __attribute__((ext_vector_type(8)));
typedef __fp16   fh2  __attribute__((ext_vector_type(2)));   // cvt_pkrtz return type
typedef float    v8f  __attribute__((ext_vector_type(8)));

#define C_CH   20
#define HW     128
#define PLANE  (HW*HW)          // 16384
#define TILE_W 32
#define TILE_H 4
#define NPIX   (TILE_W*TILE_H)  // 128 pixels / block (8 waves x 16)
#define HALO_W (TILE_W+2)       // 34
#define HALO_H (TILE_H+2)       // 6
#define NHALO  (HALO_W*HALO_H)  // 204
#define PCS    72               // perceive row stride (halfs): 60 data + pad; 144B = 36 banks -> conflict-friendly

static __device__ __forceinline__ unsigned pack2(float a, float b) {
  fh2 t = __builtin_amdgcn_cvt_pkrtz(a, b);      // v_cvt_pkrtz_f16_f32
  return __builtin_bit_cast(unsigned, t);
}
static __device__ __forceinline__ v16h load16h(const _Float16* p) {
  union { v16h v; v8h h[2]; } u;                 // two 16B LDS loads (16B-aligned addresses)
  u.h[0] = *(const v8h*)p;
  u.h[1] = *(const v8h*)(p + 8);
  return u.v;
}

// ---------------------------------------------------------------------------
// Kernel 1: pre-mask, xg, depthwise perceive, fused WMMA GEMM1+GEMM2, update.
// Writes x_mid to d_out, pre-mask + channel-3 plane of x_mid to workspace.
// ---------------------------------------------------------------------------
__global__ __launch_bounds__(256) void nca_step(
    const float* __restrict__ x, const float* __restrict__ goal,
    const float* __restrict__ rnd, const float* __restrict__ pw,
    const float* __restrict__ w1, const float* __restrict__ b1,
    const float* __restrict__ w2, float* __restrict__ out,
    float* __restrict__ ws_pre, float* __restrict__ ws_ch3)
{
  // LDS: all sizes in elements
  __shared__ __align__(16) _Float16 sA1[16*32*16]; // w1 A-fragments: 8 M-tiles x 2 K-frags
  __shared__ __align__(16) _Float16 sA2[ 8*32*16]; // w2 A-fragments: 2 M-tiles x 4 K-frags
  __shared__ float    sB1[128];
  __shared__ float    sPW[540];
  __shared__ float    sXG[C_CH*NHALO];             // xg tile + 1-halo, f32
  __shared__ __align__(16) _Float16 sPC[NPIX*PCS]; // perceive, f16, row-per-pixel

  const int tid = threadIdx.x;
  const int b   = blockIdx.z;
  const int tX0 = blockIdx.x * TILE_W;
  const int tY0 = blockIdx.y * TILE_H;
  const size_t xb = (size_t)b * C_CH * PLANE;

  // ---- marshal w1/w2 into a contiguous-K A-fragment lane layout (16x32 f16) ----
  // lane<16: V0-7 hold K=0..15 ; lane>=16: K=16..31
  for (int e = tid; e < 16*32*16; e += 256) {
    int f = e >> 9, r = e & 511;
    int lane = r >> 4, q = r & 15, v = q >> 1, hlf = q & 1;
    int m   = (f >> 1)*16 + (lane & 15);
    int kin = ((lane >= 16) ? 16 : 0) + 2*v + hlf;
    int k   = (f & 1)*32 + kin;                    // K padded 60 -> 64
    sA1[e] = (k < 60) ? (_Float16)w1[m*60 + k] : (_Float16)0.f;
  }
  for (int e = tid; e < 8*32*16; e += 256) {
    int f = e >> 9, r = e & 511;
    int lane = r >> 4, q = r & 15, v = q >> 1, hlf = q & 1;
    int m   = (f >> 2)*16 + (lane & 15);           // M padded 20 -> 32
    int kin = ((lane >= 16) ? 16 : 0) + 2*v + hlf;
    int k   = (f & 3)*32 + kin;
    sA2[e] = (m < 20) ? (_Float16)w2[m*128 + k] : (_Float16)0.f;
  }
  if (tid < 128) sB1[tid] = b1[tid];
  for (int e = tid; e < 540; e += 256) sPW[e] = pw[e];

  // ---- Stage A: pre-alive mask + xg = x + goal*mask over tile+halo ----
  for (int hp = tid; hp < NHALO; hp += 256) {
    int hx = hp % HALO_W, hy = hp / HALO_W;
    int gx = tX0 + hx - 1, gy = tY0 + hy - 1;
    bool in = ((unsigned)gx < HW) && ((unsigned)gy < HW);
    if (in) {
      float mx = -3.0e38f;
      const float* c3 = x + xb + 3*(size_t)PLANE;
      for (int dy = -1; dy <= 1; ++dy) { int yy = gy + dy; if ((unsigned)yy < HW)
        for (int dx = -1; dx <= 1; ++dx) { int xx = gx + dx; if ((unsigned)xx < HW)
          mx = fmaxf(mx, c3[yy*HW + xx]); } }
      float m = (mx > 0.1f) ? 1.f : 0.f;
      if (hx >= 1 && hx <= TILE_W && hy >= 1 && hy <= TILE_H)
        ws_pre[(size_t)b*PLANE + gy*HW + gx] = m;
      for (int c = 0; c < C_CH; ++c) {
        size_t gi = xb + (size_t)c*PLANE + gy*HW + gx;
        sXG[c*NHALO + hp] = x[gi] + m * goal[gi];
      }
    } else {
      for (int c = 0; c < C_CH; ++c) sXG[c*NHALO + hp] = 0.f;  // SAME zero-pad
    }
  }
  __syncthreads();

  // ---- Stage B: depthwise 3x3 perception -> f16 LDS [pixel][k] ----
  for (int e = tid; e < NPIX*60; e += 256) {
    int p = e / 60, o = e % 60, i = o / 3;         // group g=i feeds outputs 3g..3g+2
    int px = p & (TILE_W - 1), py = p >> 5;
    const float* g0 = &sXG[i*NHALO + py*HALO_W + px];
    const float* wv = &sPW[o*9];
    float s = 0.f;
    #pragma unroll
    for (int ky = 0; ky < 3; ++ky)
      #pragma unroll
      for (int kx = 0; kx < 3; ++kx)
        s += wv[ky*3 + kx] * g0[ky*HALO_W + kx];
    sPC[p*PCS + o] = (_Float16)s;
  }
  for (int e = tid; e < NPIX*12; e += 256) {       // zero K=60..71 pad
    int p = e / 12, j = e % 12;
    sPC[p*PCS + 60 + j] = (_Float16)0.f;
  }
  __syncthreads();

  // ---- Stage C: fused GEMM1 (128x60) -> ReLU -> GEMM2 (20x128), WMMA ----
  const int w  = tid >> 5, lane = tid & 31;
  const int hi = lane >> 4;
  const int p  = w*16 + (lane & 15);               // this lane's pixel (N column)
  // B fragments of perceive: contiguous 32B per lane
  v16h bp0 = load16h(&sPC[p*PCS +  0 + hi*16]);    // K 0..31
  v16h bp1 = load16h(&sPC[p*PCS + 32 + hi*16]);    // K 32..63 (60..63 zero)
  v8f acc20 = {}; v8f acc21 = {};

  #pragma unroll
  for (int kf = 0; kf < 4; ++kf) {                 // GEMM2 K-steps: h channels 32kf..32kf+31
    unsigned pk[2][4];
    #pragma unroll
    for (int t = 0; t < 2; ++t) {                  // GEMM1 M-tiles 2kf, 2kf+1
      int tile = 2*kf + t;
      v16h a0 = load16h(&sA1[((tile*2 + 0)*32 + lane)*16]);
      v16h a1 = load16h(&sA1[((tile*2 + 1)*32 + lane)*16]);
      v8f hacc = {};
      hacc = __builtin_amdgcn_wmma_f32_16x16x32_f16(false, a0, false, bp0, (short)0, hacc, false, false);
      hacc = __builtin_amdgcn_wmma_f32_16x16x32_f16(false, a1, false, bp1, (short)0, hacc, false, false);
      const float* bb = &sB1[tile*16 + hi*8];      // bias for this lane's 8 rows
      #pragma unroll
      for (int r = 0; r < 8; ++r) hacc[r] = fmaxf(hacc[r] + bb[r], 0.f);
      #pragma unroll
      for (int j = 0; j < 4; ++j) pk[t][j] = pack2(hacc[2*j], hacc[2*j + 1]);
    }
    // D-frag -> B-frag: same column-per-lane mapping; swap row halves across
    // lane groups with a half-wave xor shuffle instead of an LDS round-trip.
    union { v16h v; unsigned u[8]; } bf;
    #pragma unroll
    for (int j = 0; j < 4; ++j) {
      unsigned oA = __shfl_xor(pk[0][j], 16, 32);
      unsigned oB = __shfl_xor(pk[1][j], 16, 32);
      bf.u[j]     = hi ? oB : pk[0][j];            // K 0..7  / 16..23
      bf.u[4 + j] = hi ? pk[1][j] : oA;            // K 8..15 / 24..31
    }
    v16h a20 = load16h(&sA2[((0*4 + kf)*32 + lane)*16]);
    v16h a21 = load16h(&sA2[((1*4 + kf)*32 + lane)*16]);
    acc20 = __builtin_amdgcn_wmma_f32_16x16x32_f16(false, a20, false, bf.v, (short)0, acc20, false, false);
    acc21 = __builtin_amdgcn_wmma_f32_16x16x32_f16(false, a21, false, bf.v, (short)0, acc21, false, false);
  }

  // ---- Epilogue: x_mid = x + rand_mask*out ; stash channel-3 plane ----
  const int px = p & 31, py = p >> 5;
  const int gx = tX0 + px, gy = tY0 + py;
  const size_t pix = (size_t)b*PLANE + gy*HW + gx;
  const float rm = (rnd[pix] < 0.5f) ? 1.f : 0.f;
  #pragma unroll
  for (int r = 0; r < 8; ++r) {
    int c = r + 8*hi;                              // channels 0..15
    size_t gi = xb + (size_t)c*PLANE + gy*HW + gx;
    float xn = x[gi] + rm * acc20[r];
    out[gi] = xn;
    if (c == 3) ws_ch3[pix] = xn;
  }
  if (!hi) {
    #pragma unroll
    for (int r = 0; r < 4; ++r) {                  // channels 16..19
      int c = 16 + r;
      size_t gi = xb + (size_t)c*PLANE + gy*HW + gx;
      out[gi] = x[gi] + rm * acc21[r];
    }
  }
}

// ---------------------------------------------------------------------------
// Kernel 2: post-alive maxpool on x_mid channel 3, life mask, clip in place.
// ---------------------------------------------------------------------------
__global__ __launch_bounds__(256) void nca_finalize(
    const float* __restrict__ ws_pre, const float* __restrict__ ws_ch3,
    float* __restrict__ out)
{
  int gidx = blockIdx.x*256 + threadIdx.x;
  int b = gidx >> 14;
  int rem = gidx & (PLANE - 1);
  int gy = rem >> 7, gx = rem & 127;
  const float* c3 = ws_ch3 + ((size_t)b << 14);
  float mx = -3.0e38f;
  for (int dy = -1; dy <= 1; ++dy) { int yy = gy + dy; if ((unsigned)yy < HW)
    for (int dx = -1; dx <= 1; ++dx) { int xx = gx + dx; if ((unsigned)xx < HW)
      mx = fmaxf(mx, c3[yy*HW + xx]); } }
  float life = (ws_pre[gidx] > 0.5f && mx > 0.1f) ? 1.f : 0.f;
  size_t base = ((size_t)(b*C_CH) << 14) + rem;
  #pragma unroll
  for (int c = 0; c < C_CH; ++c) {
    size_t gi = base + ((size_t)c << 14);
    float v = out[gi] * life;
    out[gi] = fminf(fmaxf(v, -10.f), 10.f);
  }
}

extern "C" void kernel_launch(void* const* d_in, const int* in_sizes, int n_in,
                              void* d_out, int out_size, void* d_ws, size_t ws_size,
                              hipStream_t stream) {
  const float* x    = (const float*)d_in[0];
  const float* goal = (const float*)d_in[1];
  const float* rnd  = (const float*)d_in[2];
  const float* pw   = (const float*)d_in[3];
  const float* w1   = (const float*)d_in[4];
  const float* b1   = (const float*)d_in[5];
  const float* w2   = (const float*)d_in[6];
  float* out = (float*)d_out;
  float* ws_pre = (float*)d_ws;                    // [64*16384] pre-alive mask
  float* ws_ch3 = ws_pre + (size_t)64*PLANE;       // [64*16384] x_mid channel 3
  (void)in_sizes; (void)n_in; (void)out_size; (void)ws_size;

  dim3 g1(HW/TILE_W, HW/TILE_H, 64);               // (4, 32, 64)
  nca_step<<<g1, 256, 0, stream>>>(x, goal, rnd, pw, w1, b1, w2, out, ws_pre, ws_ch3);
  nca_finalize<<<(64*PLANE)/256, 256, 0, stream>>>(ws_pre, ws_ch3, out);
}
